// WanSelfAttention_5119601017591
// MI455X (gfx1250) — matrix operand order repaired and checked by reading, not compiled, not run on hardware
//
#include <hip/hip_runtime.h>
#include <hip/hip_bf16.h>

#define B_   2
#define L_   2048
#define DIM_ 1536
#define H_   12
#define HD_  128
#define CC_  64      // HD/2, rope channels per head
#define NTOK_ (B_ * L_)

#ifndef __has_builtin
#define __has_builtin(x) 0
#endif
#if __has_builtin(__builtin_amdgcn_tensor_load_to_lds) && __has_builtin(__builtin_amdgcn_s_wait_tensorcnt)
#define USE_TDM 1
#else
#define USE_TDM 0
#endif

typedef __attribute__((ext_vector_type(16))) _Float16 v16h;
typedef __attribute__((ext_vector_type(8)))  _Float16 v8h;
typedef __attribute__((ext_vector_type(8)))  float    v8f;
typedef __attribute__((ext_vector_type(4)))  unsigned int v4u;
typedef __attribute__((ext_vector_type(8)))  int v8i;
typedef __attribute__((ext_vector_type(4)))  int v4i;

#define WMMA_F16(a, b, c) \
  __builtin_amdgcn_wmma_f32_16x16x32_f16(false, (a), false, (b), (short)0, (c), false, false)

// ---------------------------------------------------------------------------
// Fragment loaders (wave32 layouts per CDNA5 ISA 7.12.2)
// A 16x32 f16: lane holds row m=lane&15; halves K = 8*half+[0,8) then 16+8*half+[0,8)
// B 32x16 f16: lane holds col n=lane&15; halves K = 8*half+[0,8) then 16+8*half+[0,8)
// ---------------------------------------------------------------------------
static __device__ __forceinline__ v16h frag_a(const _Float16* base, int stride, int lane, int kOff) {
  int m = lane & 15, half = lane >> 4;
  const _Float16* p = base + m * stride + kOff + 8 * half;
  v8h lo = *(const v8h*)p;
  v8h hi = *(const v8h*)(p + 16);
  return __builtin_shufflevector(lo, hi, 0,1,2,3,4,5,6,7,8,9,10,11,12,13,14,15);
}

static __device__ __forceinline__ v16h frag_b(const _Float16* baseT, int stride, int lane, int kOff) {
  int n = lane & 15, half = lane >> 4;
  const _Float16* p = baseT + n * stride + kOff + 8 * half;
  v8h lo = *(const v8h*)p;
  v8h hi = *(const v8h*)(p + 16);
  return __builtin_shufflevector(lo, hi, 0,1,2,3,4,5,6,7,8,9,10,11,12,13,14,15);
}

#if USE_TDM
// Issue a TDM 2D tile load (f16 elements) global->LDS per ISA §8 (D# groups).
// tile_w/tensor_w/row_stride in elements; pad fields per D# encoding.
static __device__ __forceinline__ void tdm_load_2d(unsigned lds_byte, const void* gptr,
                                                   unsigned tile_w, unsigned tile_h,
                                                   unsigned tensor_w, unsigned tensor_h,
                                                   unsigned row_stride,
                                                   unsigned pad_interval, unsigned pad_amount) {
  unsigned long long ga = (unsigned long long)(uintptr_t)gptr;
  v4u g0;
  g0[0] = 1u;                                             // count=1, user descriptor
  g0[1] = lds_byte;                                       // lds_addr
  g0[2] = (unsigned)(ga & 0xFFFFFFFFu);                   // global_addr[31:0]
  g0[3] = (unsigned)((ga >> 32) & 0x1FFFFFFu) | (2u << 30); // global_addr[56:32] | type=2
  v8i g1;
  g1[0] = (int)((1u << 16) | (1u << 20) | (pad_interval << 22) | (pad_amount << 25)); // 2B elems, pad_enable
  g1[1] = (int)(tensor_w << 16);                          // tensor_dim0[15:0] -> bits 63:48
  g1[2] = (int)((tensor_w >> 16) | (tensor_h << 16));     // tensor_dim0 hi / tensor_dim1 lo
  g1[3] = (int)((tensor_h >> 16) | (tile_w << 16));       // tensor_dim1 hi / tile_dim0
  g1[4] = (int)(tile_h & 0xFFFFu);                        // tile_dim1 (tile_dim2 = 0)
  g1[5] = (int)row_stride;                                // tensor_dim0_stride[31:0]
  g1[6] = 0;                                              // stride hi / dim1_stride lo
  g1[7] = 0;
  v4i gz = {0, 0, 0, 0};
#if __clang_major__ >= 23
  v8i gz8 = {0, 0, 0, 0, 0, 0, 0, 0};
  __builtin_amdgcn_tensor_load_to_lds(g0, g1, gz, gz, gz8, 0);
#else
  __builtin_amdgcn_tensor_load_to_lds(g0, g1, gz, gz, 0);
#endif
}
#endif

// ---------------------------------------------------------------------------
// GEMM (A row-major, B pre-transposed): C[M,N] = A16[M,K] * Bt16[N,K]^T + bias
// WG = 128 threads (4 waves); tile 64(M) x 128(N); wave tile 32x64 (2x2 waves).
// Register-blocked: 2 A-frags x 4 B-frags per K-substep, each B reused twice.
// Software-pipelined over double-buffered LDS (55 KB).
// ---------------------------------------------------------------------------
__global__ __launch_bounds__(128) void gemm_tn(const _Float16* __restrict__ A,
                                               const _Float16* __restrict__ Bt,
                                               const float* __restrict__ bias,
                                               float* __restrict__ C,
                                               int M, int N, int K) {
  __shared__ __align__(16) _Float16 sA[2][64 * 72];    // [m][k], stride 72 halves
  __shared__ __align__(16) _Float16 sB[2][128 * 72];   // [n][k], stride 72 halves
  const int tid = threadIdx.x, lane = tid & 31, wv = tid >> 5;
  const int wvM = wv >> 1, wvN = wv & 1;               // wave position in 2x2
  const int m0 = blockIdx.x * 64, n0 = blockIdx.y * 128;
  v8f acc[8] = {};                                     // [am*4 + bn]

  const int mA = tid >> 1, koA = (tid & 1) * 32;       // A staging: 64 rows x 64 k
  const int nB = tid;                                  // B staging: 128 rows x 64 k

  v8h rA[4], rB[8];
  auto gload = [&](int k0) {
    const _Float16* ga = A + (size_t)(m0 + mA) * K + k0 + koA;
#pragma unroll
    for (int i = 0; i < 4; ++i) rA[i] = *(const v8h*)(ga + i * 8);
    const _Float16* gb = Bt + (size_t)(n0 + nB) * K + k0;
#pragma unroll
    for (int i = 0; i < 8; ++i) rB[i] = *(const v8h*)(gb + i * 8);
  };
  auto lstore = [&](int bi) {
    _Float16* pa = &sA[bi][mA * 72 + koA];
#pragma unroll
    for (int i = 0; i < 4; ++i) *(v8h*)(pa + i * 8) = rA[i];
    _Float16* pb = &sB[bi][nB * 72];
#pragma unroll
    for (int i = 0; i < 8; ++i) *(v8h*)(pb + i * 8) = rB[i];
  };

  gload(0);
  lstore(0);
  __syncthreads();

  int buf = 0;
  for (int k0 = 0; k0 < K; k0 += 64) {
    const bool hasNext = (k0 + 64 < K);
    if (hasNext) gload(k0 + 64);      // HBM latency hidden under compute below

    const _Float16* curA = sA[buf];
    const _Float16* curB = sB[buf];
#pragma unroll
    for (int ks = 0; ks < 2; ++ks) {
      v16h a0 = frag_a(curA + (wvM * 32 +  0) * 72, 72, lane, ks * 32);
      v16h a1 = frag_a(curA + (wvM * 32 + 16) * 72, 72, lane, ks * 32);
#pragma unroll
      for (int bn = 0; bn < 4; ++bn) {
        v16h bf = frag_b(curB + (wvN * 64 + bn * 16) * 72, 72, lane, ks * 32);
        acc[bn]     = WMMA_F16(a0, bf, acc[bn]);
        acc[4 + bn] = WMMA_F16(a1, bf, acc[4 + bn]);
      }
    }
    if (hasNext) lstore(buf ^ 1);     // loadcnt wait lands here, after WMMAs
    __syncthreads();
    buf ^= 1;
  }

  const int half = lane >> 4, cn = lane & 15;
#pragma unroll
  for (int am = 0; am < 2; ++am) {
#pragma unroll
    for (int bn = 0; bn < 4; ++bn) {
      int col = n0 + wvN * 64 + bn * 16 + cn;
      float bv = bias[col];
#pragma unroll
      for (int v = 0; v < 8; ++v) {
        int row = m0 + wvM * 32 + am * 16 + v + 8 * half;
        C[(size_t)row * N + col] = acc[am * 4 + bn][v] + bv;
      }
    }
  }
}

// ---------------------------------------------------------------------------
// Flash attention: one WG (128 thr, 4 waves) per (b, head, 64-query block).
// Q fragments in registers; K tile ([key][d]) and pre-transposed V tile
// ([d][key]) staged in LDS via TDM (with manual b128 fallback); online softmax.
// Out is [b][l][dim] f32 (pre output-projection).
// ---------------------------------------------------------------------------
__global__ __launch_bounds__(128) void flash64(const _Float16* __restrict__ Q,
                                               const _Float16* __restrict__ Kmat,
                                               const _Float16* __restrict__ Vt,
                                               const int* __restrict__ seq_lens,
                                               float* __restrict__ Out) {
  __shared__ __align__(16) _Float16 sK[64 * 136];   // [key][d]  (17408 B)
  __shared__ __align__(16) _Float16 sVt[128 * 72];  // [d][key]  (18432 B)
  __shared__ __align__(16) float    sS[64 * 68];    // scores    (17408 B)
  __shared__ __align__(16) _Float16 sP[64 * 72];    // probs f16 ( 9216 B)
  __shared__ float sM[64], sL[64], sAl[64];

  const int qb = blockIdx.x, h = blockIdx.y, b = blockIdx.z;
  const int tid = threadIdx.x, lane = tid & 31, wv = tid >> 5;
  const int half = lane >> 4, cn = lane & 15;
  const size_t headOff = ((size_t)(b * H_ + h)) * L_ * HD_;  // same for Q/K and Vt
  const float scale = 0.08838834764831845f;  // 1/sqrt(128)

  // Q fragments (4 K-steps of 32) held in registers, loaded straight from HBM
  v16h qfr[4];
  {
    const _Float16* qrow = Q + headOff + (size_t)(qb * 64 + wv * 16 + cn) * HD_;
#pragma unroll
    for (int ks = 0; ks < 4; ++ks) {
      const _Float16* p = qrow + ks * 32 + 8 * half;
      v8h lo = *(const v8h*)p;
      v8h hi = *(const v8h*)(p + 16);
      qfr[ks] = __builtin_shufflevector(lo, hi, 0,1,2,3,4,5,6,7,8,9,10,11,12,13,14,15);
    }
  }
  if (tid < 64) { sM[tid] = -1e30f; sL[tid] = 0.f; }
  v8f o[8] = {};

  const int slen = seq_lens[b];
  const int nkb = (slen + 63) >> 6;

  for (int j = 0; j < nkb; ++j) {
    __syncthreads();  // previous iteration done with sK/sVt
#if USE_TDM
    // K tile: 64 rows x 128 f16, contiguous; LDS stride 136 halves ->
    //   pad 16B (pad_amount=3) after every 256B (pad_interval=5).
    // V^T tile: 128 rows x 64 f16, row stride L_; LDS stride 72 halves ->
    //   pad 16B after every 128B (pad_interval=4).
    if (wv == 0) {
      tdm_load_2d((unsigned)(uintptr_t)&sK[0],
                  Kmat + headOff + (size_t)(j * 64) * HD_,
                  HD_, 64, HD_, 64, HD_, 5, 3);
    } else if (wv == 1) {
      tdm_load_2d((unsigned)(uintptr_t)&sVt[0],
                  Vt + headOff + (size_t)(j * 64),
                  64, HD_, 64, HD_, L_, 4, 3);
    }
    __builtin_amdgcn_s_wait_tensorcnt(0);
#else
    {  // stage K tile: 64 keys x 128 d (row-major, b128 in / b128 out)
      int m = tid >> 1, off2 = (tid & 1) * 64;
      const _Float16* g = Kmat + headOff + (size_t)(j * 64 + m) * HD_ + off2;
#pragma unroll
      for (int i = 0; i < 8; ++i)
        *(v8h*)&sK[m * 136 + off2 + i * 8] = *(const v8h*)(g + i * 8);
    }
    {  // stage V^T tile from pre-transposed global: sVt[d][key]
      const _Float16* g = Vt + headOff + (size_t)tid * L_ + j * 64;
#pragma unroll
      for (int i = 0; i < 8; ++i)
        *(v8h*)&sVt[tid * 72 + i * 8] = *(const v8h*)(g + i * 8);
    }
#endif
    __syncthreads();

    // S = Q * K^T  (wave: 16 q-rows x 64 keys; K-dim = 128 in 4 steps)
    v8f sacc[4] = {};
#pragma unroll
    for (int ks = 0; ks < 4; ++ks) {
      v16h b0 = frag_b(sK + 0 * 16 * 136, 136, lane, ks * 32);
      v16h b1 = frag_b(sK + 1 * 16 * 136, 136, lane, ks * 32);
      v16h b2 = frag_b(sK + 2 * 16 * 136, 136, lane, ks * 32);
      v16h b3 = frag_b(sK + 3 * 16 * 136, 136, lane, ks * 32);
      sacc[0] = WMMA_F16(qfr[ks], b0, sacc[0]);
      sacc[1] = WMMA_F16(qfr[ks], b1, sacc[1]);
      sacc[2] = WMMA_F16(qfr[ks], b2, sacc[2]);
      sacc[3] = WMMA_F16(qfr[ks], b3, sacc[3]);
    }
    // scale + seq-len mask, spill to LDS
#pragma unroll
    for (int nt = 0; nt < 4; ++nt) {
      int col = nt * 16 + cn;
      float mask = ((j * 64 + col) < slen) ? 0.f : -1e9f;
#pragma unroll
      for (int v = 0; v < 8; ++v)
        sS[(wv * 16 + v + 8 * half) * 68 + col] = sacc[nt][v] * scale + mask;
    }
    __syncthreads();

    // online softmax: one thread per query row
    if (tid < 64) {
      const float* srow = &sS[tid * 68];
      _Float16* prow = &sP[tid * 72];
      float mOld = sM[tid], mNew = mOld;
      for (int c = 0; c < 64; ++c) mNew = fmaxf(mNew, srow[c]);
      float alpha = __expf(mOld - mNew);
      float ls = 0.f;
      for (int c = 0; c < 64; ++c) {
        float e = __expf(srow[c] - mNew);
        ls += e;
        prow[c] = (_Float16)e;
      }
      sM[tid] = mNew;
      sL[tid] = sL[tid] * alpha + ls;
      sAl[tid] = alpha;
    }
    __syncthreads();

    // rescale accumulators, then O += P * V
    float av[8];
#pragma unroll
    for (int v = 0; v < 8; ++v) av[v] = sAl[wv * 16 + v + 8 * half];
#pragma unroll
    for (int dt = 0; dt < 8; ++dt)
#pragma unroll
      for (int v = 0; v < 8; ++v) o[dt][v] *= av[v];

#pragma unroll
    for (int ks = 0; ks < 2; ++ks) {
      v16h pa = frag_a(sP + wv * 16 * 72, 72, lane, ks * 32);
#pragma unroll
      for (int g = 0; g < 2; ++g) {
        v16h b0 = frag_b(sVt + (g * 4 + 0) * 16 * 72, 72, lane, ks * 32);
        v16h b1 = frag_b(sVt + (g * 4 + 1) * 16 * 72, 72, lane, ks * 32);
        v16h b2 = frag_b(sVt + (g * 4 + 2) * 16 * 72, 72, lane, ks * 32);
        v16h b3 = frag_b(sVt + (g * 4 + 3) * 16 * 72, 72, lane, ks * 32);
        o[g * 4 + 0] = WMMA_F16(pa, b0, o[g * 4 + 0]);
        o[g * 4 + 1] = WMMA_F16(pa, b1, o[g * 4 + 1]);
        o[g * 4 + 2] = WMMA_F16(pa, b2, o[g * 4 + 2]);
        o[g * 4 + 3] = WMMA_F16(pa, b3, o[g * 4 + 3]);
      }
    }
  }
  __syncthreads();

  float linv[8];
#pragma unroll
  for (int v = 0; v < 8; ++v) linv[v] = 1.f / sL[wv * 16 + v + 8 * half];
#pragma unroll
  for (int dt = 0; dt < 8; ++dt) {
    int d = dt * 16 + cn;
#pragma unroll
    for (int v = 0; v < 8; ++v) {
      int q = qb * 64 + wv * 16 + v + 8 * half;
      Out[((size_t)b * L_ + q) * DIM_ + h * HD_ + d] = o[dt][v] * linv[v];
    }
  }
}

// ---------------------------------------------------------------------------
// Pointwise helpers
// ---------------------------------------------------------------------------
__global__ void cast_f32_f16(const float* __restrict__ in, _Float16* __restrict__ out, int n) {
  int i = blockIdx.x * blockDim.x + threadIdx.x;
  if (i < n) out[i] = (_Float16)in[i];
}

// W[k][n] f32 -> Wt[n][k] f16 (one-time; coalesced reads, scattered f16 writes)
__global__ void cast_transpose(const float* __restrict__ W, _Float16* __restrict__ Wt,
                               int rows /*K*/, int cols /*N*/) {
  int i = blockIdx.x * blockDim.x + threadIdx.x;
  if (i >= rows * cols) return;
  int n = i % cols, k = i / cols;
  Wt[(size_t)n * rows + k] = (_Float16)W[i];
}

// per-(b,pos) rope tables: pc/ps [b*L + p][64]; split 22 | 21 | 21 (C=64)
__global__ void rope_tables(const int* __restrict__ grid, const float* __restrict__ fc,
                            const float* __restrict__ fs, float* __restrict__ pc,
                            float* __restrict__ ps) {
  int p = blockIdx.x * blockDim.x + threadIdx.x;
  int b = blockIdx.y;
  int f = grid[b * 3 + 0], hh = grid[b * 3 + 1], ww = grid[b * 3 + 2];
  int hw = hh * ww;
  bool valid = p < f * hw;
  int fi = valid ? p / hw : 0;
  int hi = valid ? (p / ww) % hh : 0;
  int wi = valid ? p % ww : 0;
  size_t o = ((size_t)b * L_ + p) * CC_;
  for (int jj = 0; jj < CC_; ++jj) {
    int src = (jj < 22) ? fi : ((jj < 43) ? hi : wi);
    pc[o + jj] = valid ? fc[src * CC_ + jj] : 1.f;
    ps[o + jj] = valid ? fs[src * CC_ + jj] : 0.f;
  }
}

// fused RMS-norm (over 1536) + gain + RoPE + head-permute, f32 -> f16 [b][h][l][d]
__global__ __launch_bounds__(256) void qk_rms_rope(const float* __restrict__ xf,
                                                   const float* __restrict__ g,
                                                   const float* __restrict__ pc,
                                                   const float* __restrict__ ps,
                                                   _Float16* __restrict__ outh) {
  int row = blockIdx.x;           // b*L + l
  int b = row / L_, l = row % L_;
  const float* x = xf + (size_t)row * DIM_;
  __shared__ float red[256];
  __shared__ float s_rms;
  float ss = 0.f;
  for (int i = threadIdx.x; i < DIM_; i += 256) { float v = x[i]; ss += v * v; }
  red[threadIdx.x] = ss;
  __syncthreads();
  for (int off = 128; off > 0; off >>= 1) {
    if (threadIdx.x < (unsigned)off) red[threadIdx.x] += red[threadIdx.x + off];
    __syncthreads();
  }
  if (threadIdx.x == 0) s_rms = rsqrtf(red[0] / (float)DIM_ + 1e-6f);
  __syncthreads();
  float r = s_rms;
  for (int pi = threadIdx.x; pi < DIM_ / 2; pi += 256) {
    int h = pi / (HD_ / 2);
    int cc = pi % (HD_ / 2);
    float xr = x[2 * pi]     * r * g[2 * pi];
    float xi = x[2 * pi + 1] * r * g[2 * pi + 1];
    float c = pc[(size_t)row * CC_ + cc];
    float s = ps[(size_t)row * CC_ + cc];
    float orr = xr * c - xi * s;
    float oii = xr * s + xi * c;
    size_t o = ((size_t)(b * H_ + h) * L_ + l) * (size_t)HD_ + 2 * cc;
    outh[o] = (_Float16)orr;
    outh[o + 1] = (_Float16)oii;
  }
}

// v: f32 [b][l][h][d] -> f16 transposed [b][h][d][l]  (for direct b128 V^T staging)
__global__ void v_permute_t(const float* __restrict__ vf, _Float16* __restrict__ vt, int n) {
  int i = blockIdx.x * blockDim.x + threadIdx.x;
  if (i >= n) return;
  int d = i % HD_;
  int t = i / HD_;
  int h = t % H_;
  int t2 = t / H_;
  int l = t2 % L_;
  int b = t2 / L_;
  vt[(((size_t)(b * H_ + h)) * HD_ + d) * L_ + l] = (_Float16)vf[i];
}

// ---------------------------------------------------------------------------
extern "C" void kernel_launch(void* const* d_in, const int* in_sizes, int n_in,
                              void* d_out, int out_size, void* d_ws, size_t ws_size,
                              hipStream_t stream) {
  (void)in_sizes; (void)n_in; (void)out_size; (void)ws_size;
  const float* x          = (const float*)d_in[0];
  const int*   seq_lens   = (const int*)d_in[1];
  const int*   grid_sizes = (const int*)d_in[2];
  const float* fc         = (const float*)d_in[3];
  const float* fs         = (const float*)d_in[4];
  const float* Wq = (const float*)d_in[5];  const float* bq = (const float*)d_in[6];
  const float* Wk = (const float*)d_in[7];  const float* bk = (const float*)d_in[8];
  const float* Wv = (const float*)d_in[9];  const float* bv = (const float*)d_in[10];
  const float* Wo = (const float*)d_in[11]; const float* bo = (const float*)d_in[12];
  const float* gq = (const float*)d_in[13]; const float* gk = (const float*)d_in[14];
  float* out = (float*)d_out;

  char* ws = (char*)d_ws;
  size_t off = 0;
  auto alloc = [&](size_t bytes) -> void* {
    void* p = ws + off;
    off += (bytes + 255) & ~(size_t)255;
    return p;
  };
  const size_t tokdim = (size_t)NTOK_ * DIM_;
  _Float16* xh  = (_Float16*)alloc(tokdim * 2);
  _Float16* Wqt = (_Float16*)alloc((size_t)DIM_ * DIM_ * 2);
  _Float16* Wkt = (_Float16*)alloc((size_t)DIM_ * DIM_ * 2);
  _Float16* Wvt = (_Float16*)alloc((size_t)DIM_ * DIM_ * 2);
  _Float16* Wot = (_Float16*)alloc((size_t)DIM_ * DIM_ * 2);
  float*    qf  = (float*)alloc(tokdim * 4);
  float*    kf  = (float*)alloc(tokdim * 4);
  float*    vf  = (float*)alloc(tokdim * 4);
  _Float16* qh  = (_Float16*)alloc(tokdim * 2);
  _Float16* kh  = (_Float16*)alloc(tokdim * 2);
  _Float16* vt  = (_Float16*)alloc(tokdim * 2);
  float*    pc  = (float*)alloc((size_t)NTOK_ * CC_ * 4);
  float*    ps  = (float*)alloc((size_t)NTOK_ * CC_ * 4);
  float*    ao  = (float*)alloc(tokdim * 4);

  const int nx = (int)tokdim;
  const int nw = DIM_ * DIM_;
  cast_f32_f16<<<(nx + 255) / 256, 256, 0, stream>>>(x, xh, nx);
  cast_transpose<<<(nw + 255) / 256, 256, 0, stream>>>(Wq, Wqt, DIM_, DIM_);
  cast_transpose<<<(nw + 255) / 256, 256, 0, stream>>>(Wk, Wkt, DIM_, DIM_);
  cast_transpose<<<(nw + 255) / 256, 256, 0, stream>>>(Wv, Wvt, DIM_, DIM_);
  cast_transpose<<<(nw + 255) / 256, 256, 0, stream>>>(Wo, Wot, DIM_, DIM_);
  rope_tables<<<dim3(L_ / 256, B_), 256, 0, stream>>>(grid_sizes, fc, fs, pc, ps);

  dim3 gg(NTOK_ / 64, DIM_ / 128);
  gemm_tn<<<gg, 128, 0, stream>>>(xh, Wqt, bq, qf, NTOK_, DIM_, DIM_);
  gemm_tn<<<gg, 128, 0, stream>>>(xh, Wkt, bk, kf, NTOK_, DIM_, DIM_);
  gemm_tn<<<gg, 128, 0, stream>>>(xh, Wvt, bv, vf, NTOK_, DIM_, DIM_);

  qk_rms_rope<<<NTOK_, 256, 0, stream>>>(qf, gq, pc, ps, qh);
  qk_rms_rope<<<NTOK_, 256, 0, stream>>>(kf, gk, pc, ps, kh);
  v_permute_t<<<(nx + 255) / 256, 256, 0, stream>>>(vf, vt, nx);

  flash64<<<dim3(L_ / 64, H_, B_), 128, 0, stream>>>(qh, kh, vt, seq_lens, ao);

  cast_f32_f16<<<(nx + 255) / 256, 256, 0, stream>>>(ao, xh, nx);  // reuse xh
  gemm_tn<<<gg, 128, 0, stream>>>(xh, Wot, bo, out, NTOK_, DIM_, DIM_);
}
